// MHSAModule_57638461112573
// MI455X (gfx1250) — matrix operand order repaired and checked by reading, not compiled, not run on hardware
//
#include <hip/hip_runtime.h>
#include <math.h>

// ---------------------------------------------------------------------------
// Transformer-XL relative multi-head self-attention for MI455X (gfx1250).
// All GEMM-shaped work runs on v_wmma_f32_16x16x32_bf16 (wave32 WMMA).
// GEMM A-tiles are staged block-wide into LDS with the CDNA5 async copy
// (global_load_async_to_lds_b128 + s_wait_asynccnt).  Attention is
// flash-style: scores stay in registers, rel_shift is done with ds_bpermute
// lane gathers over a sliding band of positional WMMA tiles.
// ---------------------------------------------------------------------------

typedef __attribute__((ext_vector_type(16))) __bf16 v16bf;
typedef __attribute__((ext_vector_type(8)))  __bf16 v8bf;
typedef __attribute__((ext_vector_type(8)))  float  v8f;

#define NB 4
#define NT 1024
#define ND 512
#define NH 8
#define NDK 64
#define NL 2047
#define LPAD 2048
#define LOG2E 1.44269504088896340736f

// ------------------------------ helpers ------------------------------------

__device__ __forceinline__ v8f vzero8() {
  v8f z;
#pragma unroll
  for (int i = 0; i < 8; ++i) z[i] = 0.0f;
  return z;
}

__device__ __forceinline__ float shflf(float v, int srcLane) {
  int r = __builtin_amdgcn_ds_bpermute(srcLane << 2, __builtin_bit_cast(int, v));
  return __builtin_bit_cast(float, r);
}

__device__ __forceinline__ v16bf cat16(v8bf lo, v8bf hi) {
  return __builtin_shufflevector(lo, hi, 0, 1, 2, 3, 4, 5, 6, 7,
                                 8, 9, 10, 11, 12, 13, 14, 15);
}

// B fragment 32x16 bf16: lane n = lane&15 is the column, lanes>=16 hold
// K=16..31.  p = per-lane pointer at column-row start + k-half offset;
// 16 contiguous bf16 = two b128 loads.
__device__ __forceinline__ v16bf load_bfrag(const __bf16* p) {
  return cat16(*(const v8bf*)p, *(const v8bf*)(p + 8));
}

// A fragment 16x32 bf16: lane m = lane&15 is the row; lane<16 holds
// K 0..15, lane>=16 holds K 16..31.  rowk0 = &A[row_m][k0].
__device__ __forceinline__ v16bf load_afrag(const __bf16* rowk0, int lane) {
  int kh = lane >> 4;
  const __bf16* p = rowk0 + kh * 16;
  return cat16(*(const v8bf*)p, *(const v8bf*)(p + 8));
}

__device__ __forceinline__ v8f wmma_bf16(v16bf a, v16bf b, v8f c) {
  return __builtin_amdgcn_wmma_f32_16x16x32_bf16(false, a, false, b,
                                                 (short)0, c, false, false);
}

// CDNA5 async global->LDS copy, 16 bytes per lane (GV mode, tracked by
// ASYNCcnt).  ldsOff = LDS byte address (low 32 bits of flat shared addr).
__device__ __forceinline__ void async_g2l_b128(const void* g, unsigned ldsOff) {
  unsigned long long ga = (unsigned long long)g;
  asm volatile("global_load_async_to_lds_b128 %0, %1, off"
               :: "v"(ldsOff), "v"(ga)
               : "memory");
}
__device__ __forceinline__ void wait_async0() {
  asm volatile("s_wait_asynccnt 0x0" ::: "memory");
}

// One 16x16 score tile: (q-frag 16x64) x (64 x 16 columns of `base`),
// columns col0..col0+15, column stride NDK, clamped to maxcol.
__device__ __forceinline__ v8f score_tile(const __bf16* base, const v16bf* qf,
                                          int col0, int maxcol, int lane) {
  int n = lane & 15, kh = lane >> 4;
  int c = col0 + n;
  c = (c > maxcol) ? maxcol : c;
  const __bf16* p = base + ((size_t)c << 6) + kh * 16;
  v8f acc = vzero8();
  acc = wmma_bf16(qf[0], load_bfrag(p), acc);
  acc = wmma_bf16(qf[1], load_bfrag(p + 32), acc);
  return acc;
}

// rel_shift gather: x = pos tile at strip cols [0,16), y = [16,32).
// target (di, dj) reads strip col s = 15 - di + dj  (in [0,30]).
__device__ __forceinline__ v8f gather_shift(v8f x, v8f y, int lane) {
  int half = lane >> 4, dj = lane & 15;
  v8f t;
#pragma unroll
  for (int r = 0; r < 8; ++r) {
    int di = r + (half << 3);
    int s = 15 - di + dj;
    int src = (s & 15) + (half << 4);
    float fx = shflf(x[r], src);
    float fy = shflf(y[r], src);
    t[r] = (s < 16) ? fx : fy;
  }
  return t;
}

// ------------------------------ kernels ------------------------------------

// LayerNorm (one wave32 per row of 512) -> bf16.
__global__ __launch_bounds__(256) void xl_ln(const float* __restrict__ x,
                                             const float* __restrict__ gamma,
                                             const float* __restrict__ beta,
                                             __bf16* __restrict__ y) {
  int lane = threadIdx.x & 31;
  int wid = (blockIdx.x << 3) + (threadIdx.x >> 5);
  const float* xr = x + (size_t)wid * ND;
  float v[16], s = 0.f, ss = 0.f;
#pragma unroll
  for (int e = 0; e < 16; ++e) {
    float t = xr[lane + (e << 5)];
    v[e] = t; s += t; ss += t * t;
  }
#pragma unroll
  for (int msk = 1; msk < 32; msk <<= 1) {
    s += shflf(s, lane ^ msk);
    ss += shflf(ss, lane ^ msk);
  }
  float mean = s * (1.f / ND);
  float var = ss * (1.f / ND) - mean * mean;
  float rstd = rsqrtf(var + 1e-3f);
  __bf16* yr = y + (size_t)wid * ND;
#pragma unroll
  for (int e = 0; e < 16; ++e) {
    int c = lane + (e << 5);
    yr[c] = (__bf16)((v[e] - mean) * rstd * gamma[c] + beta[c]);
  }
}

// pos f32 -> bf16, zero-padded to 8192 rows.
__global__ __launch_bounds__(256) void xl_posbf(const float* __restrict__ pos,
                                                __bf16* __restrict__ o) {
  size_t i = (size_t)blockIdx.x * 256 + threadIdx.x;
  o[i] = (i < (size_t)NB * NL * ND) ? (__bf16)pos[i] : (__bf16)0.0f;
}

// 512x512 f32 -> transposed bf16 [N][K].
__global__ __launch_bounds__(256) void xl_tr(const float* __restrict__ in,
                                             __bf16* __restrict__ o) {
  int i = blockIdx.x * 256 + threadIdx.x;
  int r = i >> 9, c = i & 511;
  o[((size_t)c << 9) + r] = (__bf16)in[i];
}

// Generic bf16 WMMA GEMM: C[16 x 64 per wave] = A[M,512] * Wt[N,512]^T.
// The common 16x512 A strip is async-staged to LDS once per block.
// mode 0: qc/qp dual store   mode 1: K store    mode 2: V transposed store
// mode 3: P store (row guard, L mapping)        mode 4: out + bias + residual
__global__ __launch_bounds__(256) void xl_gemm(
    const __bf16* __restrict__ A, const __bf16* __restrict__ Wt,
    const float* __restrict__ bias, const float* __restrict__ cbias,
    const float* __restrict__ pbias, const float* __restrict__ xres,
    __bf16* __restrict__ out0, __bf16* __restrict__ out1,
    float* __restrict__ fout, int mode) {
  __shared__ __attribute__((aligned(16))) __bf16 atile[16 * ND];  // 16 KB
  const int lane = threadIdx.x & 31;
  const int wv = threadIdx.x >> 5;
  const int row0 = blockIdx.x << 4;
  const int n0 = wv << 6;
  const int m = lane & 15, kh = lane >> 4;

  // Async-stage the block-shared A strip (16 consecutive rows = 16 KB
  // contiguous) into LDS: 4 x b128 per thread, ASYNCcnt-tracked.
  {
    const char* gb = (const char*)(A + (size_t)row0 * ND);
    unsigned lb = (unsigned)(unsigned long long)&atile[0];
    unsigned t4 = (unsigned)threadIdx.x * 64u;
#pragma unroll
    for (int i = 0; i < 4; ++i) {
      unsigned boff = t4 + 16u * i;
      async_g2l_b128(gb + boff, lb + boff);
    }
    wait_async0();
    __syncthreads();
  }

  v8f acc[4];
#pragma unroll
  for (int c = 0; c < 4; ++c) acc[c] = vzero8();

  for (int k0 = 0; k0 < ND; k0 += 32) {
    v16bf a = load_afrag(&atile[(size_t)m * ND + k0], lane);
    const __bf16* wp = Wt + (size_t)(n0 + m) * ND + k0 + kh * 16;
    __builtin_prefetch((const void*)(wp + 64), 0, 1);  // global_prefetch_b8
#pragma unroll
    for (int c = 0; c < 4; ++c) {
      v16bf b = load_bfrag(wp + (size_t)c * 16 * ND);
      acc[c] = wmma_bf16(a, b, acc[c]);
    }
  }

#pragma unroll
  for (int c = 0; c < 4; ++c) {
#pragma unroll
    for (int r = 0; r < 8; ++r) {
      int row = row0 + r + (kh << 3);
      int ncol = n0 + (c << 4) + m;
      float v = acc[c][r];
      if (mode == 0) {
        float base = v + bias[ncol];
        int h = ncol >> 6, dk = ncol & 63;
        int bb = row >> 10, t = row & (NT - 1);
        size_t o = ((size_t)((bb * NH + h) * NT + t) << 6) + dk;
        out0[o] = (__bf16)(base + cbias[ncol]);
        out1[o] = (__bf16)(base + pbias[ncol]);
      } else if (mode == 1) {
        int h = ncol >> 6, dk = ncol & 63;
        int bb = row >> 10, t = row & (NT - 1);
        out0[((size_t)((bb * NH + h) * NT + t) << 6) + dk] =
            (__bf16)(v + bias[ncol]);
      } else if (mode == 2) {
        int h = ncol >> 6, dk = ncol & 63;
        int bb = row >> 10, t = row & (NT - 1);
        out0[((size_t)((bb * NH + h) * NDK + dk) << 10) + t] =
            (__bf16)(v + bias[ncol]);
      } else if (mode == 3) {
        if (row < NB * NL) {
          int bb = row / NL, l = row - bb * NL;
          int h = ncol >> 6, dk = ncol & 63;
          out0[((size_t)((bb * NH + h) * LPAD + l) << 6) + dk] = (__bf16)v;
        }
      } else {
        size_t o = (size_t)row * ND + ncol;
        fout[o] = v + bias[ncol] + xres[o];
      }
    }
  }
}

// Flash attention: one wave per (b, h, 16-query tile), 32-key blocks.
__global__ __launch_bounds__(128) void xl_attn(
    const __bf16* __restrict__ qc, const __bf16* __restrict__ qp,
    const __bf16* __restrict__ kk, const __bf16* __restrict__ vt,
    const __bf16* __restrict__ pp, __bf16* __restrict__ ob) {
  __shared__ __attribute__((aligned(16))) __bf16 plds[4][16 * 32];
  const int lane = threadIdx.x & 31;
  const int wv = threadIdx.x >> 5;
  const int bh = blockIdx.y;
  const int b = bh >> 3, h = bh & 7;
  const int i0 = ((blockIdx.x << 2) + wv) << 4;
  const int m = lane & 15, kh = lane >> 4;
  const int half = kh, dj = m;

  const __bf16* qcb = qc + ((size_t)bh * NT + i0 + m) * NDK;
  const __bf16* qpb = qp + ((size_t)bh * NT + i0 + m) * NDK;
  const __bf16* kb = kk + (size_t)bh * NT * NDK;
  const __bf16* vb = vt + (size_t)bh * NDK * NT;
  const __bf16* pb = pp + (size_t)bh * LPAD * NDK;
  __bf16* lw = &plds[wv][0];

  v16bf qcf[2], qpf[2];
#pragma unroll
  for (int ks = 0; ks < 2; ++ks) {
    qcf[ks] = load_afrag(qcb + ks * 32, lane);
    qpf[ks] = load_afrag(qpb + ks * 32, lane);
  }

  v8f oacc[4];
#pragma unroll
  for (int c = 0; c < 4; ++c) oacc[c] = vzero8();
  float rmax[8], rsum[8];
#pragma unroll
  for (int r = 0; r < 8; ++r) { rmax[r] = -3.0e38f; rsum[r] = 0.0f; }

  const int cb0 = NT - i0 - 15;  // strip base: m = T - i + j, s = 15-di+dj
  v8f pa = score_tile(pb, qpf, cb0, NL - 1, lane);

  for (int j0 = 0; j0 < NT; j0 += 32) {
    v8f pbt = score_tile(pb, qpf, cb0 + j0 + 16, NL - 1, lane);
    v8f pct = score_tile(pb, qpf, cb0 + j0 + 32, NL - 1, lane);
    v8f c0t = score_tile(kb, qcf, j0, NT - 1, lane);
    v8f c1t = score_tile(kb, qcf, j0 + 16, NT - 1, lane);
    v8f sp0 = gather_shift(pa, pbt, lane);
    v8f sp1 = gather_shift(pbt, pct, lane);

#pragma unroll
    for (int r = 0; r < 8; ++r) {
      float s0 = (c0t[r] + sp0[r]) * 0.125f;  // 1/sqrt(DK)
      float s1 = (c1t[r] + sp1[r]) * 0.125f;
      float mx = fmaxf(s0, s1);
#pragma unroll
      for (int msk = 1; msk < 16; msk <<= 1) mx = fmaxf(mx, shflf(mx, lane ^ msk));
      float nm = fmaxf(rmax[r], mx);
      float sc = exp2f((rmax[r] - nm) * LOG2E);
      rmax[r] = nm;
      float p0 = exp2f((s0 - nm) * LOG2E);
      float p1 = exp2f((s1 - nm) * LOG2E);
      float ps = p0 + p1;
#pragma unroll
      for (int msk = 1; msk < 16; msk <<= 1) ps += shflf(ps, lane ^ msk);
      rsum[r] = rsum[r] * sc + ps;
#pragma unroll
      for (int c = 0; c < 4; ++c) oacc[c][r] *= sc;
      int di = r + (half << 3);
      lw[di * 32 + dj] = (__bf16)p0;
      lw[di * 32 + 16 + dj] = (__bf16)p1;
    }

    // C-layout -> A-layout transpose for the 16x32 P tile via LDS.
    asm volatile("s_wait_dscnt 0x0" ::: "memory");
    v16bf pf = load_afrag(lw + m * 32, lane);
#pragma unroll
    for (int c = 0; c < 4; ++c) {
      const __bf16* vp = vb + (size_t)((c << 4) + m) * NT + j0 + kh * 16;
      v16bf bf = load_bfrag(vp);
      oacc[c] = wmma_bf16(pf, bf, oacc[c]);
    }
    pa = pct;  // trailing pos tile becomes leading tile of next block
  }

#pragma unroll
  for (int r = 0; r < 8; ++r) {
    float inv = 1.0f / rsum[r];
    int row = i0 + r + (half << 3);
    size_t obase = ((size_t)(b * NT + row) * ND) + (h << 6) + dj;
#pragma unroll
    for (int c = 0; c < 4; ++c) ob[obase + (c << 4)] = (__bf16)(oacc[c][r] * inv);
  }
}

// ------------------------------ launcher -----------------------------------

extern "C" void kernel_launch(void* const* d_in, const int* in_sizes, int n_in,
                              void* d_out, int out_size, void* d_ws,
                              size_t ws_size, hipStream_t stream) {
  (void)in_sizes; (void)n_in; (void)out_size; (void)ws_size;
  const float* x     = (const float*)d_in[0];
  const float* pos   = (const float*)d_in[1];
  const float* cbias = (const float*)d_in[2];
  const float* pbias = (const float*)d_in[3];
  const float* gamma = (const float*)d_in[4];
  const float* beta  = (const float*)d_in[5];
  const float* Wq    = (const float*)d_in[6];
  const float* bq    = (const float*)d_in[7];
  const float* Wk    = (const float*)d_in[8];
  const float* bk    = (const float*)d_in[9];
  const float* Wv    = (const float*)d_in[10];
  const float* bv    = (const float*)d_in[11];
  const float* Wp    = (const float*)d_in[12];
  const float* Wo    = (const float*)d_in[13];
  const float* bo    = (const float*)d_in[14];
  float* out = (float*)d_out;

  char* w = (char*)d_ws;
  auto alloc = [&](size_t elems) {
    char* p = w;
    w += ((elems * sizeof(__bf16)) + 255) & ~(size_t)255;
    return (__bf16*)p;
  };
  __bf16* ybf  = alloc((size_t)NB * NT * ND);          // 4096 x 512
  __bf16* pzb  = alloc((size_t)8192 * ND);             // padded pos
  __bf16* WqT  = alloc((size_t)ND * ND);
  __bf16* WkT  = alloc((size_t)ND * ND);
  __bf16* WvT  = alloc((size_t)ND * ND);
  __bf16* WpT  = alloc((size_t)ND * ND);
  __bf16* WoT  = alloc((size_t)ND * ND);
  __bf16* qcbf = alloc((size_t)NB * NH * NT * NDK);
  __bf16* qpbf = alloc((size_t)NB * NH * NT * NDK);
  __bf16* kbf  = alloc((size_t)NB * NH * NT * NDK);
  __bf16* vtbf = alloc((size_t)NB * NH * NDK * NT);
  __bf16* pbf  = alloc((size_t)NB * NH * LPAD * NDK);
  __bf16* obf  = alloc((size_t)NB * NT * ND);

  xl_ln<<<512, 256, 0, stream>>>(x, gamma, beta, ybf);
  xl_posbf<<<16384, 256, 0, stream>>>(pos, pzb);
  xl_tr<<<1024, 256, 0, stream>>>(Wq, WqT);
  xl_tr<<<1024, 256, 0, stream>>>(Wk, WkT);
  xl_tr<<<1024, 256, 0, stream>>>(Wv, WvT);
  xl_tr<<<1024, 256, 0, stream>>>(Wp, WpT);
  xl_tr<<<1024, 256, 0, stream>>>(Wo, WoT);

  xl_gemm<<<256, 256, 0, stream>>>(ybf, WqT, bq, cbias, pbias, nullptr,
                                   qcbf, qpbf, nullptr, 0);
  xl_gemm<<<256, 256, 0, stream>>>(ybf, WkT, bk, nullptr, nullptr, nullptr,
                                   kbf, nullptr, nullptr, 1);
  xl_gemm<<<256, 256, 0, stream>>>(ybf, WvT, bv, nullptr, nullptr, nullptr,
                                   vtbf, nullptr, nullptr, 2);
  xl_gemm<<<512, 256, 0, stream>>>(pzb, WpT, nullptr, nullptr, nullptr, nullptr,
                                   pbf, nullptr, nullptr, 3);

  xl_attn<<<dim3(16, 32), 128, 0, stream>>>(qcbf, qpbf, kbf, vtbf, pbf, obf);

  xl_gemm<<<256, 256, 0, stream>>>(obf, WoT, bo, nullptr, nullptr, x,
                                   nullptr, nullptr, out, 4);
}
